// LSTMModel_86655260164285
// MI455X (gfx1250) — matrix operand order repaired and checked by reading, not compiled, not run on hardware
//
#include <hip/hip_runtime.h>
#include <hip/hip_bf16.h>

#define B_TOT   1024
#define T_LEN   512
#define EMB     32
#define HID     64
#define GATES   256     // 4*HID
#define VOCAB   32000

typedef __attribute__((ext_vector_type(16))) __bf16 v16bf;
typedef __attribute__((ext_vector_type(8)))  float  v8f;

union Frag { v16bf v; unsigned u[8]; };

__device__ __forceinline__ unsigned short f2bf(float f) {
    union { float f; unsigned u; } a; a.f = f;
    unsigned u = a.u;
    unsigned r = u + 0x7FFFu + ((u >> 16) & 1u);   // round-to-nearest-even
    return (unsigned short)(r >> 16);
}
__device__ __forceinline__ unsigned pack2(float lo, float hi) {
    return (unsigned)f2bf(lo) | ((unsigned)f2bf(hi) << 16);
}

// Fast transcendentals on CDNA5 TRANS pipe
__device__ __forceinline__ float tanh_fast(float x) {
#if __has_builtin(__builtin_amdgcn_tanhf)
    return __builtin_amdgcn_tanhf(x);              // single v_tanh_f32
#else
    return 2.0f * __builtin_amdgcn_rcpf(1.0f + __expf(-2.0f * x)) - 1.0f;
#endif
}
__device__ __forceinline__ float sigm(float x) {
    return fmaf(0.5f, tanh_fast(0.5f * x), 0.5f);  // v_mul + v_tanh + v_fma
}

// A-fragment (16x32 bf16) K base for VGPR v / lane, per ISA layout table:
// lanes 0-15: v0-3 -> K=2v..2v+1 ; v4-7 -> K=16+2(v-4) ; lanes 16-31: +8
__device__ __forceinline__ int kbaseA(int v, bool hiHalf) {
    int k = (v < 4) ? (2 * v) : (16 + 2 * (v - 4));
    return k + (hiHalf ? 8 : 0);
}
// B-fragment (32x16 bf16): same K order as A (v0-3 -> K=2v, v4-7 -> K=16+2(v-4); lanes 16-31: +8)
__device__ __forceinline__ int kbaseB(int v, bool hiHalf) {
    return kbaseA(v, hiHalf);
}

// ---------------------------------------------------------------------------
// Kernel 1: fused embedding-gather + input GEMM + LSTM recurrence.
// One block = 16 batch rows, 8 waves, each wave owns 2 gate N-tiles (32 cols).
// Token/embedding loads are software-pipelined one step ahead so the two
// dependent L2 accesses overlap the barrier + WMMA + elementwise phases.
// ---------------------------------------------------------------------------
__global__ __launch_bounds__(256) void lstm_recurrence_kernel(
    const int*   __restrict__ x,      // [B,T]
    const float* __restrict__ emb,    // [VOCAB,EMB]
    const float* __restrict__ W_ih,   // [GATES,EMB]
    const float* __restrict__ W_hh,   // [GATES,HID]
    const float* __restrict__ b_ih,   // [GATES]
    const float* __restrict__ b_hh,   // [GATES]
    unsigned short* __restrict__ hws) // [B,HID] bf16 bits (out)
{
    __shared__ __align__(16) unsigned short eA[16 * EMB];   // step embeddings, bf16
    __shared__ __align__(16) unsigned short hA[16 * HID];   // hidden state, bf16
    __shared__ __align__(16) float gates[16 * GATES];       // raw gates, f32

    const int tid  = threadIdx.x;
    const int lane = tid & 31;
    const int w    = tid >> 5;           // wave 0..7
    const int lo16 = lane & 15;
    const bool hiHalf = lane >= 16;
    const int b0 = blockIdx.x * 16;

    // ---- one-time: weights as persistent bf16 B-fragments, bias per lane ----
    Frag wih[2];          // input GEMM, K=32 (one frag per tile)
    Frag whh[2][2];       // hidden GEMM, K=64 (two K-frags per tile)
    float bias[2];
    #pragma unroll
    for (int j = 0; j < 2; ++j) {
        const int n = w * 32 + j * 16 + lo16;    // gate column this lane owns
        bias[j] = b_ih[n] + b_hh[n];
        #pragma unroll
        for (int v = 0; v < 8; ++v) {
            const int k = kbaseB(v, hiHalf);
            wih[j].u[v] = pack2(W_ih[n * EMB + k], W_ih[n * EMB + k + 1]);
            #pragma unroll
            for (int kb = 0; kb < 2; ++kb) {
                const int kk = kb * 32 + k;
                whh[j][kb].u[v] = pack2(W_hh[n * HID + kk], W_hh[n * HID + kk + 1]);
            }
        }
    }

    // ---- init h = 0 (LDS), c = 0 (regs) ----
    {
        unsigned* hz = (unsigned*)hA;            // 512 dwords
        hz[tid]       = 0u;
        hz[tid + 256] = 0u;
    }
    float c_reg[4] = {0.f, 0.f, 0.f, 0.f};

    // gather assignment: thread handles 2 consecutive emb cols of one row
    const int gm   = tid >> 4;                   // row 0..15
    const int gcol = (tid & 15) * 2;             // col 0..30 (even)

    // ---- pipeline prologue: stage embeddings for t = 0 ----
    int tokN = x[(b0 + gm) * T_LEN + 0];
    float e0 = emb[tokN * EMB + gcol];
    float e1 = emb[tokN * EMB + gcol + 1];

    for (int t = 0; t < T_LEN; ++t) {
        // ---- P1: publish staged embeddings for step t ----
        *(unsigned*)&eA[gm * EMB + gcol] = pack2(e0, e1);
        // issue token-index load for step t+1 (latency hidden by bar1+WMMA)
        const int tn = (t + 1 < T_LEN) ? (t + 1) : t;
        tokN = x[(b0 + gm) * T_LEN + tn];
        __syncthreads();   // eA ready; hA (prev step) ready

        // ---- P2: WMMA  gates = e*W_ih^T + h*W_hh^T + bias ----
        Frag ea, ha0, ha1;
        #pragma unroll
        for (int v = 0; v < 8; ++v) {
            const int kA = kbaseA(v, hiHalf);
            ea.u[v]  = *(const unsigned*)&eA[lo16 * EMB + kA];
            ha0.u[v] = *(const unsigned*)&hA[lo16 * HID + kA];
            ha1.u[v] = *(const unsigned*)&hA[lo16 * HID + 32 + kA];
        }
        #pragma unroll
        for (int j = 0; j < 2; ++j) {
            v8f acc;
            #pragma unroll
            for (int r = 0; r < 8; ++r) acc[r] = bias[j];
            acc = __builtin_amdgcn_wmma_f32_16x16x32_bf16(false, ea.v,  false, wih[j].v,    (short)0, acc, false, false);
            acc = __builtin_amdgcn_wmma_f32_16x16x32_bf16(false, ha0.v, false, whh[j][0].v, (short)0, acc, false, false);
            acc = __builtin_amdgcn_wmma_f32_16x16x32_bf16(false, ha1.v, false, whh[j][1].v, (short)0, acc, false, false);
            // ---- P3: spill raw gate tile to LDS (C layout: M = r + 8*hi) ----
            const int n = w * 32 + j * 16 + lo16;
            const int Mb = hiHalf ? 8 : 0;
            #pragma unroll
            for (int r = 0; r < 8; ++r)
                gates[(Mb + r) * GATES + n] = acc[r];
        }
        __syncthreads();   // gates ready; WMMA reads of eA/hA done

        // issue embedding-row load for step t+1 (latency hidden by P4)
        e0 = emb[tokN * EMB + gcol];
        e1 = emb[tokN * EMB + gcol + 1];

        // ---- P4: elementwise LSTM cell update (v_tanh fast path) ----
        #pragma unroll
        for (int k = 0; k < 4; ++k) {
            const int idx = tid * 4 + k;
            const int m   = idx >> 6;
            const int col = idx & 63;
            const float iv = sigm(gates[m * GATES + col]);
            const float fv = sigm(gates[m * GATES + 64 + col]);
            const float gv = tanh_fast(gates[m * GATES + 128 + col]);
            const float ov = sigm(gates[m * GATES + 192 + col]);
            c_reg[k] = fv * c_reg[k] + iv * gv;
            const float hv = ov * tanh_fast(c_reg[k]);
            hA[m * HID + col] = f2bf(hv);
        }
        // next iteration's barrier after P1 orders hA writes vs. WMMA reads
    }

    __syncthreads();
    // ---- export final h (bf16) for the FC kernel ----
    {
        const unsigned* src = (const unsigned*)hA;
        unsigned* dst = (unsigned*)(hws + (size_t)b0 * HID);
        dst[tid]       = src[tid];
        dst[tid + 256] = src[tid + 256];
    }
}

// ---------------------------------------------------------------------------
// Kernel 2: out = h_last @ fc_W^T + fc_b   ([1024,32000], K=64, bf16 WMMA)
// grid (250, 64), 8 waves/block, one 16x16 tile per wave. Output-BW bound.
// ---------------------------------------------------------------------------
__global__ __launch_bounds__(256) void lstm_fc_kernel(
    const unsigned short* __restrict__ hws,  // [B,HID] bf16 bits
    const float* __restrict__ fc_W,          // [VOCAB,HID]
    const float* __restrict__ fc_b,          // [VOCAB]
    float* __restrict__ out)                 // [B,VOCAB]
{
    const int tid  = threadIdx.x;
    const int lane = tid & 31;
    const int w    = tid >> 5;
    const int lo16 = lane & 15;
    const bool hiHalf = lane >= 16;

    const int m0 = blockIdx.y * 16;
    const int n0 = (blockIdx.x * 8 + w) * 16;
    const int N  = n0 + lo16;

    // A fragments: final hidden rows m0..m0+15, K=64 as two K=32 frags
    Frag a0, a1, bw0, bw1;
    #pragma unroll
    for (int v = 0; v < 8; ++v) {
        const int kA = kbaseA(v, hiHalf);
        a0.u[v] = *(const unsigned*)&hws[(size_t)(m0 + lo16) * HID + kA];
        a1.u[v] = *(const unsigned*)&hws[(size_t)(m0 + lo16) * HID + 32 + kA];
        const int kB = kbaseB(v, hiHalf);
        bw0.u[v] = pack2(fc_W[(size_t)N * HID + kB],      fc_W[(size_t)N * HID + kB + 1]);
        bw1.u[v] = pack2(fc_W[(size_t)N * HID + 32 + kB], fc_W[(size_t)N * HID + 32 + kB + 1]);
    }

    v8f acc;
    const float bias = fc_b[N];
    #pragma unroll
    for (int r = 0; r < 8; ++r) acc[r] = bias;

    acc = __builtin_amdgcn_wmma_f32_16x16x32_bf16(false, a0.v, false, bw0.v, (short)0, acc, false, false);
    acc = __builtin_amdgcn_wmma_f32_16x16x32_bf16(false, a1.v, false, bw1.v, (short)0, acc, false, false);

    const int Mb = hiHalf ? 8 : 0;
    #pragma unroll
    for (int r = 0; r < 8; ++r)
        out[(size_t)(m0 + Mb + r) * VOCAB + N] = acc[r];
}

extern "C" void kernel_launch(void* const* d_in, const int* in_sizes, int n_in,
                              void* d_out, int out_size, void* d_ws, size_t ws_size,
                              hipStream_t stream) {
    (void)in_sizes; (void)n_in; (void)out_size; (void)ws_size;
    const int*   x    = (const int*)  d_in[0];
    const float* emb  = (const float*)d_in[1];
    const float* W_ih = (const float*)d_in[2];
    const float* W_hh = (const float*)d_in[3];
    const float* b_ih = (const float*)d_in[4];
    const float* b_hh = (const float*)d_in[5];
    const float* fc_W = (const float*)d_in[6];
    const float* fc_b = (const float*)d_in[7];

    unsigned short* hws = (unsigned short*)d_ws;   // [B,HID] bf16 bits (128 KB)

    lstm_recurrence_kernel<<<B_TOT / 16, 256, 0, stream>>>(
        x, emb, W_ih, W_hh, b_ih, b_hh, hws);

    dim3 grid(VOCAB / 16 / 8, B_TOT / 16);         // (250, 64)
    lstm_fc_kernel<<<grid, 256, 0, stream>>>(hws, fc_W, fc_b, (float*)d_out);
}
